// GATNet_85452669321923
// MI455X (gfx1250) — matrix operand order repaired and checked by reading, not compiled, not run on hardware
//
#include <hip/hip_runtime.h>

#define N_NODES 6000
#define F_IN    300
#define HID     128
#define NHEAD   8
#define ALPHA   0.2f

#define MT   (N_NODES / 16)          // 375 row tiles
#define KC1  (320 / 32)              // 10 (F_IN padded 300->320)
#define KCH  (1024 / 32)             // 32 (H*HID)
#define KCV  ((N_NODES + 31) / 32)   // 188 (neighbor dim padded to 6016)
#define WPB  3                       // waves per block (375 = 3*125)

typedef __attribute__((ext_vector_type(16))) __bf16 v16bf;
typedef __attribute__((ext_vector_type(8)))  float  v8f;
typedef __attribute__((ext_vector_type(4)))  float  f32x4;

union BFrag { v16bf v; __bf16 e[16]; };
union FAcc  { v8f   v; float  e[8];  };

// Scheduler grouping hints (no-op if builtin unavailable).
// masks: 8 = MFMA/WMMA, 32 = VMEM read, 256 = DS read
#if defined(__has_builtin)
#if __has_builtin(__builtin_amdgcn_sched_group_barrier)
#define SCHED_GROUP(m, s, i) __builtin_amdgcn_sched_group_barrier((m), (s), (i))
#endif
#endif
#ifndef SCHED_GROUP
#define SCHED_GROUP(m, s, i)
#endif

// ---------------------------------------------------------------------------
// Fragment packing (layouts per ISA §7.12.2).
// A-fragment (16x32 bf16): lane L holds row M=L&15, K = (j&8?16:0)+(L>>4)*8+(j&7)
// B-fragment (32x16 bf16): lane L holds col N=L&15, K = (j&8?16:0)+(L>>4)*8+(j&7)
// Packed so each lane's 16 bf16 are contiguous (one 32B vector load):
//   A: ((rowTile*KC + kc)*32 + lane)*16 + j
//   B: (((head*KC + kc)*8 + nt)*32 + lane)*16 + j
// ---------------------------------------------------------------------------

__global__ void pack_a_f32(const float* __restrict__ src, __bf16* __restrict__ dst,
                           int mTiles, int KC, int Ksrc, int ldsrc) {
    int t = blockIdx.x * blockDim.x + threadIdx.x;
    if (t >= mTiles * KC * 32) return;
    int lane = t & 31;
    int kc   = (t >> 5) % KC;
    int rt   = t / (KC * 32);
    int row  = rt * 16 + (lane & 15);
    int kh   = lane >> 4;
    __bf16* d = dst + (size_t)t * 16;
#pragma unroll
    for (int j = 0; j < 16; ++j) {
        int K = kc * 32 + ((j & 8) ? 16 : 0) + kh * 8 + (j & 7);
        float v = (K < Ksrc) ? src[(size_t)row * ldsrc + K] : 0.0f;
        d[j] = (__bf16)v;
    }
}

__global__ void pack_a_bf16(const __bf16* __restrict__ src, __bf16* __restrict__ dst,
                            int mTiles, int KC, int Ksrc, int ldsrc) {
    int t = blockIdx.x * blockDim.x + threadIdx.x;
    if (t >= mTiles * KC * 32) return;
    int lane = t & 31;
    int kc   = (t >> 5) % KC;
    int rt   = t / (KC * 32);
    int row  = rt * 16 + (lane & 15);
    int kh   = lane >> 4;
    __bf16* d = dst + (size_t)t * 16;
#pragma unroll
    for (int j = 0; j < 16; ++j) {
        int K = kc * 32 + ((j & 8) ? 16 : 0) + kh * 8 + (j & 7);
        d[j] = (K < Ksrc) ? src[(size_t)row * ldsrc + K] : (__bf16)0.0f;
    }
}

// src: per-head row-major Ksrc x 128 fp32; K >= Ksrc up to KC*32 packed as zeros
__global__ void pack_b_f32(const float* __restrict__ src, __bf16* __restrict__ dst,
                           int KC, int Ksrc, size_t srcHeadStride, int nHeads) {
    int t = blockIdx.x * blockDim.x + threadIdx.x;
    if (t >= nHeads * KC * 8 * 32) return;
    int lane = t & 31;
    int nt   = (t >> 5) & 7;
    int kc   = (t >> 8) % KC;
    int h    = t / (KC * 8 * 32);
    int col  = nt * 16 + (lane & 15);
    int kg   = lane >> 4;
    const float* s = src + (size_t)h * srcHeadStride;
    __bf16* d = dst + (size_t)t * 16;
#pragma unroll
    for (int j = 0; j < 16; ++j) {
        int K = kc * 32 + ((j & 8) ? 16 : 0) + kg * 8 + (j & 7);
        float v = (K < Ksrc) ? s[(size_t)K * HID + col] : 0.0f;
        d[j] = (__bf16)v;
    }
}

// ---------------------------------------------------------------------------
// Generic bf16 WMMA GEMM: out[h] (Mx128 fp32) = A (MxK) @ B[h] (Kx128)
// One wave per (rowTile, head). Scheduling hint groups the 18 b128 operand
// loads of a k-chunk into one clause followed by 8 back-to-back WMMAs.
// ---------------------------------------------------------------------------
__global__ void __launch_bounds__(32 * WPB)
gemm_wmma(const __bf16* __restrict__ Ap, const __bf16* __restrict__ Bp,
          float* __restrict__ out, int mTiles, int KC, size_t outHeadStride) {
    int lane = threadIdx.x & 31;
    int wave = threadIdx.x >> 5;
    int rt = blockIdx.x * WPB + wave;
    int h  = blockIdx.y;
    if (rt >= mTiles) return;

    FAcc acc[8] = {};
    const __bf16* apBase = Ap + (size_t)rt * KC * 32 * 16;
    const __bf16* bpBase = Bp + (size_t)h * KC * 8 * 32 * 16;

    for (int kc = 0; kc < KC; ++kc) {
        v16bf a = *(const v16bf*)(apBase + ((size_t)kc * 32 + lane) * 16);
        const __bf16* bp = bpBase + (size_t)kc * 8 * 32 * 16 + (size_t)lane * 16;
        v16bf b[8];
#pragma unroll
        for (int nt = 0; nt < 8; ++nt)
            b[nt] = *(const v16bf*)(bp + (size_t)nt * 32 * 16);
#pragma unroll
        for (int nt = 0; nt < 8; ++nt)
            acc[nt].v = __builtin_amdgcn_wmma_f32_16x16x32_bf16(
                false, a, false, b[nt], (short)0, acc[nt].v, false, false);
        SCHED_GROUP(32, 18, 0);   // A (2xb128) + 8 B fragments (16xb128)
        SCHED_GROUP(8,   8, 0);   // then 8 WMMAs back-to-back
    }
    int colN = lane & 15;
    int mOff = (lane >> 4) * 8;
    float* o = out + (size_t)h * outHeadStride + (size_t)rt * 16 * HID;
#pragma unroll
    for (int nt = 0; nt < 8; ++nt)
#pragma unroll
        for (int r = 0; r < 8; ++r)
            o[(size_t)(r + mOff) * HID + nt * 16 + colN] = acc[nt].e[r];
}

// f1[h][n] = Wh[h][n]·a[h][0:128], f2 = ·a[h][128:256]
__global__ void dot_f(const float* __restrict__ Wh, const float* __restrict__ aVec,
                      float* __restrict__ f1, float* __restrict__ f2,
                      int nRows, int nHeads, size_t whHeadStride, int aHeadStride) {
    int t = blockIdx.x * blockDim.x + threadIdx.x;
    if (t >= nRows * nHeads) return;
    int n = t % nRows;
    int h = t / nRows;
    const float* w  = Wh + (size_t)h * whHeadStride + (size_t)n * HID;
    const float* a1 = aVec + (size_t)h * aHeadStride;
    const float* a2 = a1 + HID;
    float s1 = 0.f, s2 = 0.f;
#pragma unroll 4
    for (int k = 0; k < HID; ++k) { float v = w[k]; s1 += v * a1[k]; s2 += v * a2[k]; }
    f1[t] = s1;
    f2[t] = s2;
}

// ---------------------------------------------------------------------------
// Fused masked-softmax attention + (attn @ V) via WMMA.
// One wave per (rowTile, head); 3 waves per block, all live (375 = 3*125).
// V fragments (identical for all waves of a block) are double-buffer staged
// through LDS once per block. adj/f2 probed with vectorized float4 loads;
// bounds handling statically peeled into the final (tail) chunk.
// ---------------------------------------------------------------------------
__device__ inline void acc_max4(float myF1, f32x4 ad, f32x4 fv, float& mx) {
#pragma unroll
    for (int i = 0; i < 4; ++i)
        if (ad[i] > 0.0f) {
            float e = myF1 + fv[i];
            e = (e > 0.0f) ? e : ALPHA * e;
            mx = fmaxf(mx, e);
        }
}

__device__ inline float acc_p4(float myF1, f32x4 ad, f32x4 fv, float mx, __bf16* dst) {
    float s = 0.0f;
#pragma unroll
    for (int i = 0; i < 4; ++i) {
        float p = 0.0f;
        if (ad[i] > 0.0f) {
            float e = myF1 + fv[i];
            e = (e > 0.0f) ? e : ALPHA * e;
            p = __expf(e - mx);
        }
        s += p;
        dst[i] = (__bf16)p;
    }
    return s;
}

template <bool ELU, typename OutT>
__global__ void __launch_bounds__(32 * WPB)
gat_attn(const float* __restrict__ f1, const float* __restrict__ f2,
         const float* __restrict__ adj, const __bf16* __restrict__ Vp,
         OutT* __restrict__ out, size_t outRowStride, int headColMul) {
    __shared__ alignas(32) __bf16 vbuf[2][8 * 32 * 16];   // 2 x 8KB

    int tid  = threadIdx.x;
    int lane = tid & 31;
    int wave = tid >> 5;
    int rt = blockIdx.x * WPB + wave;     // always < MT (375 = WPB*gridDim.x)
    int h  = blockIdx.y;

    int row = rt * 16 + (lane & 15);
    int kh  = lane >> 4;
    const float* f2h    = f2 + (size_t)h * N_NODES;
    const float* adjRow = adj + (size_t)row * N_NODES;
    float myF1 = f1[(size_t)h * N_NODES + row];

    // ---- pass 1: row max of masked leaky_relu(f1+f2); 187 full chunks + tail
    float mx = -3.0e38f;
    for (int kc = 0; kc < KCV - 1; ++kc) {
        int b0 = kc * 32 + kh * 8;
        f32x4 a0 = *(const f32x4*)(adjRow + b0);
        f32x4 a1 = *(const f32x4*)(adjRow + b0 + 4);
        f32x4 a2 = *(const f32x4*)(adjRow + b0 + 16);
        f32x4 a3 = *(const f32x4*)(adjRow + b0 + 20);
        f32x4 q0 = *(const f32x4*)(f2h + b0);
        f32x4 q1 = *(const f32x4*)(f2h + b0 + 4);
        f32x4 q2 = *(const f32x4*)(f2h + b0 + 16);
        f32x4 q3 = *(const f32x4*)(f2h + b0 + 20);
        acc_max4(myF1, a0, q0, mx);
        acc_max4(myF1, a1, q1, mx);
        acc_max4(myF1, a2, q2, mx);
        acc_max4(myF1, a3, q3, mx);
    }
    {   // tail chunk: low half (m = 5984+kh*8 .. +7) statically valid, high half OOB
        int b0 = (KCV - 1) * 32 + kh * 8;
        f32x4 a0 = *(const f32x4*)(adjRow + b0);
        f32x4 a1 = *(const f32x4*)(adjRow + b0 + 4);
        f32x4 q0 = *(const f32x4*)(f2h + b0);
        f32x4 q1 = *(const f32x4*)(f2h + b0 + 4);
        acc_max4(myF1, a0, q0, mx);
        acc_max4(myF1, a1, q1, mx);
    }
    mx = fmaxf(mx, __shfl_xor(mx, 16, 32));   // lanes L and L+16 share a row

    // ---- pass 2: p = exp(e-mx) in A-fragment layout; p @ V with LDS-staged V
    const __bf16* vpHead = Vp + (size_t)h * KCV * 8 * 32 * 16;
    auto stage = [&](int buf, int kc) {
        const v16bf* s = (const v16bf*)(vpHead + (size_t)kc * 8 * 32 * 16);
        v16bf* d = (v16bf*)vbuf[buf];
        for (int i = tid; i < 256; i += 32 * WPB) d[i] = s[i];
    };

    FAcc acc[8] = {};
    float sum = 0.0f;
    stage(0, 0);
    __syncthreads();
    for (int kc = 0; kc < KCV; ++kc) {
        if (kc + 1 < KCV) stage((kc + 1) & 1, kc + 1);

        BFrag af;
        int b0 = kc * 32 + kh * 8;
        f32x4 a0 = *(const f32x4*)(adjRow + b0);
        f32x4 a1 = *(const f32x4*)(adjRow + b0 + 4);
        f32x4 q0 = *(const f32x4*)(f2h + b0);
        f32x4 q1 = *(const f32x4*)(f2h + b0 + 4);
        sum += acc_p4(myF1, a0, q0, mx, af.e + 0);
        sum += acc_p4(myF1, a1, q1, mx, af.e + 4);
        if (kc < KCV - 1) {
            f32x4 a2 = *(const f32x4*)(adjRow + b0 + 16);
            f32x4 a3 = *(const f32x4*)(adjRow + b0 + 20);
            f32x4 q2 = *(const f32x4*)(f2h + b0 + 16);
            f32x4 q3 = *(const f32x4*)(f2h + b0 + 20);
            sum += acc_p4(myF1, a2, q2, mx, af.e + 8);
            sum += acc_p4(myF1, a3, q3, mx, af.e + 12);
        } else {
#pragma unroll
            for (int j = 8; j < 16; ++j) af.e[j] = (__bf16)0.0f;
        }

        const __bf16* bp = vbuf[kc & 1] + (size_t)lane * 16;
        v16bf b[8];
#pragma unroll
        for (int nt = 0; nt < 8; ++nt)
            b[nt] = *(const v16bf*)(bp + (size_t)nt * 32 * 16);
#pragma unroll
        for (int nt = 0; nt < 8; ++nt)
            acc[nt].v = __builtin_amdgcn_wmma_f32_16x16x32_bf16(
                false, af.v, false, b[nt], (short)0, acc[nt].v, false, false);
        SCHED_GROUP(256, 16, 0);  // 8 B fragments from LDS (16 x ds_load_b128)
        SCHED_GROUP(8,    8, 0);  // then 8 WMMAs back-to-back

        __syncthreads();   // staged buffer ready / compute buffer reusable
    }
    sum += __shfl_xor(sum, 16, 32);           // full row sum (lanes L / L+16)

    // ---- epilogue: normalize, optional ELU, store
    int colN = lane & 15;
    int mOff = (lane >> 4) * 8;
    OutT* o = out + (size_t)rt * 16 * outRowStride + (size_t)h * headColMul;
#pragma unroll
    for (int r = 0; r < 8; ++r) {
        float rs = __shfl(sum, r + mOff, 32); // lane M holds the sum of row M
        float inv = 1.0f / rs;
#pragma unroll
        for (int nt = 0; nt < 8; ++nt) {
            float v = acc[nt].e[r] * inv;
            if (ELU) v = (v > 0.0f) ? v : (__expf(v) - 1.0f);
            o[(size_t)(r + mOff) * outRowStride + nt * 16 + colN] = (OutT)v;
        }
    }
}

// ---------------------------------------------------------------------------
extern "C" void kernel_launch(void* const* d_in, const int* in_sizes, int n_in,
                              void* d_out, int out_size, void* d_ws, size_t ws_size,
                              hipStream_t stream) {
    const float* x       = (const float*)d_in[0];
    const float* adj     = (const float*)d_in[1];
    const float* W_heads = (const float*)d_in[2];
    const float* a_heads = (const float*)d_in[3];
    const float* W_out   = (const float*)d_in[4];
    const float* a_out   = (const float*)d_in[5];
    float* out = (float*)d_out;
    (void)in_sizes; (void)n_in; (void)out_size; (void)ws_size;

    char* ws = (char*)d_ws;
    size_t off = 0;
    auto alloc = [&](size_t bytes) -> void* {
        void* p = ws + off;
        off = (off + bytes + 255) & ~(size_t)255;
        return p;
    };
    __bf16* Apx  = (__bf16*)alloc((size_t)MT * KC1 * 32 * 16 * 2);
    __bf16* BpW  = (__bf16*)alloc((size_t)NHEAD * KC1 * 8 * 32 * 16 * 2);
    float*  Wh   = (float*) alloc((size_t)NHEAD * N_NODES * HID * 4);
    float*  f1v  = (float*) alloc((size_t)NHEAD * N_NODES * 4);
    float*  f2v  = (float*) alloc((size_t)NHEAD * N_NODES * 4);
    __bf16* Vp1  = (__bf16*)alloc((size_t)NHEAD * KCV * 8 * 32 * 16 * 2);
    __bf16* hcat = (__bf16*)alloc((size_t)N_NODES * NHEAD * HID * 2);
    __bf16* Aph  = (__bf16*)alloc((size_t)MT * KCH * 32 * 16 * 2);
    __bf16* BpO  = (__bf16*)alloc((size_t)KCH * 8 * 32 * 16 * 2);
    float*  Wh2  = (float*) alloc((size_t)N_NODES * HID * 4);
    float*  g1v  = (float*) alloc((size_t)N_NODES * 4);
    float*  g2v  = (float*) alloc((size_t)N_NODES * 4);
    __bf16* Vp2  = (__bf16*)alloc((size_t)KCV * 8 * 32 * 16 * 2);

    dim3 blk(256);
    // 1) pack x into A fragments (K padded 300->320)
    { int tot = MT * KC1 * 32;
      pack_a_f32<<<dim3((tot + 255) / 256), blk, 0, stream>>>(x, Apx, MT, KC1, F_IN, F_IN); }
    // 2) pack W_heads into B fragments
    { int tot = NHEAD * KC1 * 8 * 32;
      pack_b_f32<<<dim3((tot + 255) / 256), blk, 0, stream>>>(W_heads, BpW, KC1, F_IN,
                                                              (size_t)F_IN * HID, NHEAD); }
    // 3) Wh = x @ W_heads   (8 heads)
    gemm_wmma<<<dim3(MT / WPB, NHEAD), dim3(32 * WPB), 0, stream>>>(
        Apx, BpW, Wh, MT, KC1, (size_t)N_NODES * HID);
    // 4) f1/f2 attention logits
    { int tot = NHEAD * N_NODES;
      dot_f<<<dim3((tot + 255) / 256), blk, 0, stream>>>(Wh, a_heads, f1v, f2v,
                                                         N_NODES, NHEAD, (size_t)N_NODES * HID, 2 * HID); }
    // 5) pack Wh into V fragments (rows padded 6000->6016)
    { int tot = NHEAD * KCV * 8 * 32;
      pack_b_f32<<<dim3((tot + 255) / 256), blk, 0, stream>>>(Wh, Vp1, KCV, N_NODES,
                                                              (size_t)N_NODES * HID, NHEAD); }
    // 6) layer-1 fused masked softmax + attn@Wh + ELU -> hcat (bf16)
    gat_attn<true, __bf16><<<dim3(MT / WPB, NHEAD), dim3(32 * WPB), 0, stream>>>(
        f1v, f2v, adj, Vp1, hcat, (size_t)(NHEAD * HID), HID);
    // 7) pack hcat into A fragments
    { int tot = MT * KCH * 32;
      pack_a_bf16<<<dim3((tot + 255) / 256), blk, 0, stream>>>(hcat, Aph, MT, KCH,
                                                               NHEAD * HID, NHEAD * HID); }
    // 8) pack W_out into B fragments
    { int tot = KCH * 8 * 32;
      pack_b_f32<<<dim3((tot + 255) / 256), blk, 0, stream>>>(W_out, BpO, KCH, NHEAD * HID, 0, 1); }
    // 9) Wh2 = hcat @ W_out
    gemm_wmma<<<dim3(MT / WPB, 1), dim3(32 * WPB), 0, stream>>>(Aph, BpO, Wh2, MT, KCH, 0);
    // 10) g1/g2 logits
    { int tot = N_NODES;
      dot_f<<<dim3((tot + 255) / 256), blk, 0, stream>>>(Wh2, a_out, g1v, g2v, N_NODES, 1, 0, 2 * HID); }
    // 11) pack Wh2 into V fragments
    { int tot = KCV * 8 * 32;
      pack_b_f32<<<dim3((tot + 255) / 256), blk, 0, stream>>>(Wh2, Vp2, KCV, N_NODES, 0, 1); }
    // 12) layer-2 fused attention -> final fp32 output
    gat_attn<false, float><<<dim3(MT / WPB, 1), dim3(32 * WPB), 0, stream>>>(
        g1v, g2v, adj, Vp2, out, HID, 0);
}
